// SpatialTemporalGNN_57982058496340
// MI455X (gfx1250) — matrix operand order repaired and checked by reading, not compiled, not run on hardware
//
#include <hip/hip_runtime.h>
#include <math.h>

// ---------------- problem constants (from reference) ----------------
#define T_STEPS 16
#define N_NODES 20
#define BATCH   4096
#define FDIM    64
#define OUTD    2
#define BNODES  (BATCH * N_NODES)   // 81920
#define NEDGE   327680
#define DD1     (FDIM * N_NODES)    // 1280
#define DD2     (DD1 / 2)           // 640
#define DD3     (DD1 / 4)           // 320
#define KCAT    (3 * FDIM)          // 192 : K-concatenated Chebyshev basis

typedef _Float16 h16;
typedef __attribute__((ext_vector_type(2)))  _Float16 v2h;
typedef __attribute__((ext_vector_type(16))) _Float16 v16h;
typedef __attribute__((ext_vector_type(8)))  float    v8f;

// ---------------- small utility kernels ----------------

__global__ void fill_zero_kernel(float* __restrict__ p, long n) {
    long i = (long)blockIdx.x * blockDim.x + threadIdx.x;
    if (i < n) p[i] = 0.0f;
}

__global__ void deg_accum_kernel(const int* __restrict__ src,
                                 const float* __restrict__ w,
                                 float* __restrict__ deg, int nE) {
    int e = blockIdx.x * blockDim.x + threadIdx.x;
    if (e < nE) atomicAdd(deg + src[e], w[e]);
}

__global__ void edge_norm_kernel(const int* __restrict__ src,
                                 const int* __restrict__ dst,
                                 const float* __restrict__ w,
                                 const float* __restrict__ deg,
                                 float* __restrict__ nw, int nE) {
    int e = blockIdx.x * blockDim.x + threadIdx.x;
    if (e >= nE) return;
    float ds = deg[src[e]], dd = deg[dst[e]];
    float is = ds > 0.0f ? rsqrtf(ds) : 0.0f;
    float id = dd > 0.0f ? rsqrtf(dd) : 0.0f;
    nw[e] = -is * w[e] * id;
}

// Tx[:,0:64]=S ; Tx[:,64:128]=0 ; Tx[:,128:192]=-S   (Tx2 prefill = -Tx0)
__global__ void init_tx_kernel(float* __restrict__ Tx,
                               const float* __restrict__ S, long n) {
    long i = (long)blockIdx.x * blockDim.x + threadIdx.x;
    if (i >= n) return;
    long r = i >> 6;
    int  c = (int)(i & 63);
    float v = S[i];
    float* row = Tx + r * (long)KCAT;
    row[c]            = v;
    row[FDIM + c]     = 0.0f;
    row[2 * FDIM + c] = -v;
}

// out[dst[e], c..c+3] += scale*nw[e] * in[src[e], c..c+3]
__global__ void scatter_prop_kernel(const float* __restrict__ in, int ldin,
                                    float* __restrict__ out, int ldout,
                                    const int* __restrict__ src,
                                    const int* __restrict__ dst,
                                    const float* __restrict__ nw,
                                    float scale, int nE) {
    int tid = blockIdx.x * blockDim.x + threadIdx.x;
    int e = tid >> 4;
    if (e >= nE) return;
    int c = (tid & 15) << 2;
    int s = src[e], d = dst[e];
    float w = scale * nw[e];
    const float4 x = *(const float4*)(in + (long)s * ldin + c);
    float* o = out + (long)d * ldout + c;
    atomicAdd(o + 0, w * x.x);
    atomicAdd(o + 1, w * x.y);
    atomicAdd(o + 2, w * x.z);
    atomicAdd(o + 3, w * x.w);
}

__global__ void ewise_mul_kernel(float* __restrict__ r,
                                 const float* __restrict__ h, long n) {
    long i = (long)blockIdx.x * blockDim.x + threadIdx.x;
    if (i < n) r[i] = h[i] * r[i];
}

// H = relu(Z*H + (1-Z)*Ht)
__global__ void gru_update_kernel(float* __restrict__ H,
                                  const float* __restrict__ Z,
                                  const float* __restrict__ Ht, long n) {
    long i = (long)blockIdx.x * blockDim.x + threadIdx.x;
    if (i >= n) return;
    float z = Z[i];
    float v = z * H[i] + (1.0f - z) * Ht[i];
    H[i] = fmaxf(v, 0.0f);
}

__global__ void softmax2_kernel(float* __restrict__ out,
                                const float* __restrict__ in, int rows) {
    int r = blockIdx.x * blockDim.x + threadIdx.x;
    if (r >= rows) return;
    float a = in[2 * r], b = in[2 * r + 1];
    float m = fmaxf(a, b);
    float ea = expf(a - m), eb = expf(b - m);
    float s = 1.0f / (ea + eb);
    out[2 * r] = ea * s;
    out[2 * r + 1] = eb * s;
}

// ---------------- fragment packing (f32 -> f16 in WMMA lane layout) ----------
// A fragment (16x32, ISA 7.12.2): lane&15 = M row; kg=(lane>>4)*8;
//   half[2v]=A[row][kf(v)], half[2v+1]=A[row][kf(v)+1], kf(v)= v<4 ? kg+2v : 16+kg+2(v-4)
// Apack layout: [(mtile*kSteps + ks)*32 + lane][16] halves (32B/lane, contiguous)
__global__ void pack_a_kernel(const float* __restrict__ A, int lda,
                              h16* __restrict__ Ap, unsigned mTiles, unsigned kSteps) {
    unsigned i = blockIdx.x * blockDim.x + threadIdx.x;   // one thread = 2 halves
    unsigned total = mTiles * kSteps * 256u;
    if (i >= total) return;
    unsigned v    = i & 7u;
    unsigned lane = (i >> 3) & 31u;
    unsigned fs   = i >> 8;                 // mtile*kSteps + ks
    unsigned mtile = fs / kSteps;
    unsigned ks   = fs - mtile * kSteps;
    unsigned row  = mtile * 16u + (lane & 15u);
    unsigned kg   = (lane >> 4) * 8u;
    unsigned k    = ks * 32u + ((v < 4u) ? (kg + 2u * v) : (16u + kg + 2u * (v - 4u)));
    float2 x = *(const float2*)(A + (long)row * lda + k);
    v2h o; o[0] = (h16)x.x; o[1] = (h16)x.y;
    *(v2h*)(Ap + ((size_t)(fs * 32u + lane) << 4) + 2u * v) = o;
}

// B fragment (32x16): col = lane&15; kg=(lane>>4)*8; half[j]=W[k0+(j<8 ? kg+j : 16+kg+j-8)][col]
// Bpack layout: [(ks*ntiles + nt)*32 + lane][16] halves
__global__ void pack_b_kernel(const float* __restrict__ W, int N, unsigned nTiles,
                              h16* __restrict__ Bp, unsigned kSteps) {
    unsigned i = blockIdx.x * blockDim.x + threadIdx.x;   // one thread = 1 half
    unsigned total = kSteps * nTiles * 512u;
    if (i >= total) return;
    unsigned j    = i & 15u;
    unsigned lane = (i >> 4) & 31u;
    unsigned fs   = i >> 9;                 // ks*nTiles + nt
    unsigned ks   = fs / nTiles;
    unsigned nt   = fs - ks * nTiles;
    int col = (int)(nt * 16u + (lane & 15u));
    unsigned k = ks * 32u + ((j < 8u) ? ((lane >> 4) * 8u + j) : (16u + (lane >> 4) * 8u + (j - 8u)));
    float v = (col < N) ? W[(size_t)k * N + col] : 0.0f;
    Bp[(size_t)(fs * 32u + lane) * 16u + j] = (h16)v;
}

// ---------------- packed WMMA GEMM (CDNA5 wave32) ----------------
// out[M,N] = act( A1@W1 + b1 [+ A2@W2 + b2] [+ addend] )
// Apack/Bpack are fragment-packed f16; each wave owns a 32x64 super-tile
// (2 m-tiles x 4 n-tiles): per k-step 2 A-loads + 4 B-loads feed 8 WMMAs.
// ~130 VGPRs live; launch_bounds(256,4) budgets 256 VGPRs/lane (no spills).
__global__ void __launch_bounds__(256, 4)
wmma_gemm_kernel(const h16* __restrict__ Ap1,
                 const h16* __restrict__ Bp1,
                 const float* __restrict__ b1,
                 const h16* __restrict__ Ap2,
                 const h16* __restrict__ Bp2,
                 const float* __restrict__ b2,
                 const float* __restrict__ addend, int ldadd,
                 float* __restrict__ out, int ldout,
                 int M, int N, int kSteps, int nTiles, int act) {
    int gtid   = blockIdx.x * blockDim.x + threadIdx.x;
    int wave   = gtid >> 5;
    int lane   = threadIdx.x & 31;
    int nStrip = (N + 63) >> 6;
    int mGrp   = M >> 5;                     // M must be a multiple of 32
    if (wave >= mGrp * nStrip) return;       // wave-uniform exit (EXEC stays full)
    long mt0 = (long)(wave / nStrip) * 2;    // first of 2 m-tiles
    int  nt0 = (wave % nStrip) * 4;          // first of 4 n-tiles

    v8f acc[2][4];
#pragma unroll
    for (int i = 0; i < 2; ++i)
#pragma unroll
        for (int t = 0; t < 4; ++t) acc[i][t] = (v8f){0, 0, 0, 0, 0, 0, 0, 0};

    const long aStride = (long)kSteps * 32;  // v16h elems between m-tiles
    const long bStep   = (long)nTiles * 32;  // v16h elems between k-steps

    if (nt0 + 4 <= nTiles) {
        // ---- full-strip fast path: branch-free
#pragma unroll 1
        for (int pass = 0; pass < 2; ++pass) {
            const v16h* Af = (const v16h*)(pass ? Ap2 : Ap1);
            const v16h* Bf = (const v16h*)(pass ? Bp2 : Bp1);
            if (Af == nullptr) continue;     // uniform scalar branch
            Af += mt0 * aStride + lane;
            Bf += (long)nt0 * 32 + lane;
#pragma unroll 1
            for (int ks = 0; ks < kSteps; ++ks) {
                const v16h* ak = Af + (long)ks * 32;
                const v16h* bk = Bf + (long)ks * bStep;
                v16h a0  = ak[0];
                v16h a1  = ak[aStride];
                v16h bf0 = bk[0];
                v16h bf1 = bk[32];
                v16h bf2 = bk[64];
                v16h bf3 = bk[96];
                acc[0][0] = __builtin_amdgcn_wmma_f32_16x16x32_f16(
                    false, a0, false, bf0, (short)0, acc[0][0], false, false);
                acc[1][0] = __builtin_amdgcn_wmma_f32_16x16x32_f16(
                    false, a1, false, bf0, (short)0, acc[1][0], false, false);
                acc[0][1] = __builtin_amdgcn_wmma_f32_16x16x32_f16(
                    false, a0, false, bf1, (short)0, acc[0][1], false, false);
                acc[1][1] = __builtin_amdgcn_wmma_f32_16x16x32_f16(
                    false, a1, false, bf1, (short)0, acc[1][1], false, false);
                acc[0][2] = __builtin_amdgcn_wmma_f32_16x16x32_f16(
                    false, a0, false, bf2, (short)0, acc[0][2], false, false);
                acc[1][2] = __builtin_amdgcn_wmma_f32_16x16x32_f16(
                    false, a1, false, bf2, (short)0, acc[1][2], false, false);
                acc[0][3] = __builtin_amdgcn_wmma_f32_16x16x32_f16(
                    false, a0, false, bf3, (short)0, acc[0][3], false, false);
                acc[1][3] = __builtin_amdgcn_wmma_f32_16x16x32_f16(
                    false, a1, false, bf3, (short)0, acc[1][3], false, false);
            }
        }
    } else {
        // ---- tail strip (only the final N=2 layer lands here)
#pragma unroll 1
        for (int pass = 0; pass < 2; ++pass) {
            const v16h* Af = (const v16h*)(pass ? Ap2 : Ap1);
            const v16h* Bf = (const v16h*)(pass ? Bp2 : Bp1);
            if (Af == nullptr) continue;
            Af += mt0 * aStride + lane;
            Bf += (long)nt0 * 32 + lane;
#pragma unroll 1
            for (int ks = 0; ks < kSteps; ++ks) {
                const v16h* ak = Af + (long)ks * 32;
                v16h a0 = ak[0];
                v16h a1 = ak[aStride];
#pragma unroll
                for (int t = 0; t < 4; ++t) {
                    if (nt0 + t >= nTiles) break;   // wave-uniform
                    v16h bf = Bf[(long)ks * bStep + t * 32];
                    acc[0][t] = __builtin_amdgcn_wmma_f32_16x16x32_f16(
                        false, a0, false, bf, (short)0, acc[0][t], false, false);
                    acc[1][t] = __builtin_amdgcn_wmma_f32_16x16x32_f16(
                        false, a1, false, bf, (short)0, acc[1][t], false, false);
                }
            }
        }
    }

    int rl = lane >> 4;   // 0/1 -> M offset 0/8
    int nl = lane & 15;
#pragma unroll
    for (int i = 0; i < 2; ++i) {
#pragma unroll
        for (int t = 0; t < 4; ++t) {
            int col = nt0 * 16 + t * 16 + nl;
            if (col >= N) continue;
            float bb = (b1 ? b1[col] : 0.0f) + (b2 ? b2[col] : 0.0f);
#pragma unroll
            for (int r = 0; r < 8; ++r) {
                long row = (mt0 + i) * 16 + r + 8 * rl;
                float v = acc[i][t][r] + bb;
                if (addend) v += addend[row * ldadd + col];
                if (act == 1)      v = fmaxf(v, 0.0f);
                else if (act == 2) v = 1.0f / (1.0f + expf(-v));
                else if (act == 3) v = tanhf(v);
                out[row * ldout + col] = v;
            }
        }
    }
}

// ---------------- host-side launch helpers ----------------

static inline void launch_fill0(float* p, long n, hipStream_t s) {
    fill_zero_kernel<<<(unsigned)((n + 255) / 256), 256, 0, s>>>(p, n);
}

static inline void launch_pack_a(const float* A, int lda, h16* Ap, int M, int K,
                                 hipStream_t s) {
    long total = (long)(M / 16) * (K / 32) * 256;
    pack_a_kernel<<<(unsigned)((total + 255) / 256), 256, 0, s>>>(
        A, lda, Ap, (unsigned)(M / 16), (unsigned)(K / 32));
}

static inline void launch_pack_b(const float* W, int N, h16* Bp, int K, hipStream_t s) {
    int nTiles = (N + 15) / 16;
    long total = (long)(K / 32) * nTiles * 512;
    pack_b_kernel<<<(unsigned)((total + 255) / 256), 256, 0, s>>>(
        W, N, (unsigned)nTiles, Bp, (unsigned)(K / 32));
}

static inline void launch_gemm(const h16* Ap1, const h16* Bp1, const float* b1,
                               const h16* Ap2, const h16* Bp2, const float* b2,
                               const float* addend, int ldadd,
                               float* out, int ldout, int M, int N, int K, int act,
                               hipStream_t s) {
    int nStrip = (N + 63) / 64;
    long thr = (long)(M / 32) * nStrip * 32;   // one wave per 32x64 super-tile
    wmma_gemm_kernel<<<(unsigned)((thr + 255) / 256), 256, 0, s>>>(
        Ap1, Bp1, b1, Ap2, Bp2, b2, addend, ldadd, out, ldout,
        M, N, K / 32, (N + 15) / 16, act);
}

extern "C" void kernel_launch(void* const* d_in, const int* in_sizes, int n_in,
                              void* d_out, int out_size, void* d_ws, size_t ws_size,
                              hipStream_t stream) {
    (void)in_sizes; (void)n_in; (void)out_size; (void)ws_size;

    const float* x_temporal  = (const float*)d_in[0];
    const int*   edge_index  = (const int*)d_in[1];
    const float* edge_weight = (const float*)d_in[2];
    const float* W_xz = (const float*)d_in[3];  const float* b_xz = (const float*)d_in[4];
    const float* W_hz = (const float*)d_in[5];  const float* b_hz = (const float*)d_in[6];
    const float* W_xr = (const float*)d_in[7];  const float* b_xr = (const float*)d_in[8];
    const float* W_hr = (const float*)d_in[9];  const float* b_hr = (const float*)d_in[10];
    const float* W_xh = (const float*)d_in[11]; const float* b_xh = (const float*)d_in[12];
    const float* W_hh = (const float*)d_in[13]; const float* b_hh = (const float*)d_in[14];
    const float* W1 = (const float*)d_in[15];   const float* b1 = (const float*)d_in[16];
    const float* W2 = (const float*)d_in[17];   const float* b2 = (const float*)d_in[18];
    const float* W3 = (const float*)d_in[19];   const float* b3 = (const float*)d_in[20];

    const int* src = edge_index;
    const int* dst = edge_index + NEDGE;

    // workspace carve-out (256B aligned slots, counted in floats)
    float* ws = (float*)d_ws;
    size_t off = 0;
    auto alloc = [&](size_t nfloats) {
        float* p = ws + off; off += (nfloats + 63) & ~(size_t)63; return p;
    };
    auto allocH = [&](size_t nhalves) { return (h16*)alloc((nhalves + 1) / 2); };

    float* TxX = alloc((size_t)BNODES * KCAT);
    float* TxH = alloc((size_t)BNODES * KCAT);
    float* Zb  = alloc((size_t)BNODES * FDIM);
    float* Rb  = alloc((size_t)BNODES * FDIM);
    float* XHb = alloc((size_t)BNODES * FDIM);
    float* Hb  = alloc((size_t)BNODES * FDIM);
    float* deg = alloc((size_t)BNODES);
    float* nw  = alloc((size_t)NEDGE);
    float* t1  = alloc((size_t)BATCH * DD2);
    float* t2  = alloc((size_t)BATCH * DD3);
    float* t3  = alloc((size_t)BATCH * OUTD);
    h16* PaX = allocH((size_t)BNODES * KCAT);     // also reused for MLP A-packs
    h16* PaH = allocH((size_t)BNODES * KCAT);
    h16* Bxz = allocH((size_t)KCAT * FDIM);
    h16* Bhz = allocH((size_t)KCAT * FDIM);
    h16* Bxr = allocH((size_t)KCAT * FDIM);
    h16* Bhr = allocH((size_t)KCAT * FDIM);
    h16* Bxh = allocH((size_t)KCAT * FDIM);
    h16* Bhh = allocH((size_t)KCAT * FDIM);
    h16* Bw1 = allocH((size_t)DD1 * DD2);
    h16* Bw2 = allocH((size_t)DD2 * DD3);
    h16* Bw3 = allocH((size_t)DD3 * 16);

    const long nBF = (long)BNODES * FDIM;
    const int  eblk = (NEDGE + 255) / 256;
    const int  sblk = (NEDGE * 16 + 255) / 256;
    const unsigned pblk = (unsigned)((nBF + 255) / 256);

    // scaled-Laplacian edge weights
    launch_fill0(deg, BNODES, stream);
    deg_accum_kernel<<<eblk, 256, 0, stream>>>(src, edge_weight, deg, NEDGE);
    edge_norm_kernel<<<eblk, 256, 0, stream>>>(src, dst, edge_weight, deg, nw, NEDGE);
    launch_fill0(Hb, nBF, stream);

    // pack all weights to f16 fragments (once per launch)
    launch_pack_b(W_xz, FDIM, Bxz, KCAT, stream);
    launch_pack_b(W_hz, FDIM, Bhz, KCAT, stream);
    launch_pack_b(W_xr, FDIM, Bxr, KCAT, stream);
    launch_pack_b(W_hr, FDIM, Bhr, KCAT, stream);
    launch_pack_b(W_xh, FDIM, Bxh, KCAT, stream);
    launch_pack_b(W_hh, FDIM, Bhh, KCAT, stream);
    launch_pack_b(W1, DD2, Bw1, DD1, stream);
    launch_pack_b(W2, DD3, Bw2, DD2, stream);
    launch_pack_b(W3, OUTD, Bw3, DD3, stream);

    for (int t = 0; t < T_STEPS; ++t) {
        const float* Xt = x_temporal + (size_t)t * nBF;

        // Chebyshev basis of X: Tx0=X, Tx1=prop(X), Tx2=2*prop(Tx1)-X
        init_tx_kernel<<<pblk, 256, 0, stream>>>(TxX, Xt, nBF);
        scatter_prop_kernel<<<sblk, 256, 0, stream>>>(Xt, FDIM, TxX + FDIM, KCAT,
                                                      src, dst, nw, 1.0f, NEDGE);
        scatter_prop_kernel<<<sblk, 256, 0, stream>>>(TxX + FDIM, KCAT, TxX + 2 * FDIM, KCAT,
                                                      src, dst, nw, 2.0f, NEDGE);
        launch_pack_a(TxX, KCAT, PaX, BNODES, KCAT, stream);

        // Chebyshev basis of H
        init_tx_kernel<<<pblk, 256, 0, stream>>>(TxH, Hb, nBF);
        scatter_prop_kernel<<<sblk, 256, 0, stream>>>(Hb, FDIM, TxH + FDIM, KCAT,
                                                      src, dst, nw, 1.0f, NEDGE);
        scatter_prop_kernel<<<sblk, 256, 0, stream>>>(TxH + FDIM, KCAT, TxH + 2 * FDIM, KCAT,
                                                      src, dst, nw, 2.0f, NEDGE);
        launch_pack_a(TxH, KCAT, PaH, BNODES, KCAT, stream);

        // Z = sigmoid(chebX@Wxz + chebH@Whz + biases)
        launch_gemm(PaX, Bxz, b_xz, PaH, Bhz, b_hz, nullptr, 0,
                    Zb, FDIM, BNODES, FDIM, KCAT, 2, stream);
        // R = sigmoid(chebX@Wxr + chebH@Whr + biases)
        launch_gemm(PaX, Bxr, b_xr, PaH, Bhr, b_hr, nullptr, 0,
                    Rb, FDIM, BNODES, FDIM, KCAT, 2, stream);
        // XH = chebX@Wxh + bxh (linear, saved for Ht)
        launch_gemm(PaX, Bxh, b_xh, nullptr, nullptr, nullptr, nullptr, 0,
                    XHb, FDIM, BNODES, FDIM, KCAT, 0, stream);

        // HR = H * R  (in place on R)
        ewise_mul_kernel<<<pblk, 256, 0, stream>>>(Rb, Hb, nBF);

        // Chebyshev basis of H*R (reuse TxH/PaH)
        init_tx_kernel<<<pblk, 256, 0, stream>>>(TxH, Rb, nBF);
        scatter_prop_kernel<<<sblk, 256, 0, stream>>>(Rb, FDIM, TxH + FDIM, KCAT,
                                                      src, dst, nw, 1.0f, NEDGE);
        scatter_prop_kernel<<<sblk, 256, 0, stream>>>(TxH + FDIM, KCAT, TxH + 2 * FDIM, KCAT,
                                                      src, dst, nw, 2.0f, NEDGE);
        launch_pack_a(TxH, KCAT, PaH, BNODES, KCAT, stream);

        // Ht = tanh(chebHR@Whh + bhh + XH)  -> into Rb
        launch_gemm(PaH, Bhh, b_hh, nullptr, nullptr, nullptr, XHb, FDIM,
                    Rb, FDIM, BNODES, FDIM, KCAT, 3, stream);

        // H = relu(Z*H + (1-Z)*Ht)
        gru_update_kernel<<<pblk, 256, 0, stream>>>(Hb, Zb, Rb, nBF);
    }

    // MLP head: H reshaped [4096,1280] is contiguous
    launch_pack_a(Hb, DD1, PaX, BATCH, DD1, stream);
    launch_gemm(PaX, Bw1, b1, nullptr, nullptr, nullptr, nullptr, 0,
                t1, DD2, BATCH, DD2, DD1, 1, stream);
    launch_pack_a(t1, DD2, PaX, BATCH, DD2, stream);
    launch_gemm(PaX, Bw2, b2, nullptr, nullptr, nullptr, nullptr, 0,
                t2, DD3, BATCH, DD3, DD2, 1, stream);
    launch_pack_a(t2, DD3, PaX, BATCH, DD3, stream);
    launch_gemm(PaX, Bw3, b3, nullptr, nullptr, nullptr, nullptr, 0,
                t3, OUTD, BATCH, OUTD, DD3, 0, stream);
    softmax2_kernel<<<(BATCH + 255) / 256, 256, 0, stream>>>((float*)d_out, t3, BATCH);
}
